// Codebook_62620623175879
// MI455X (gfx1250) — compile-verified, matrix operand order checked by reading, not run on hardware
//
#include <hip/hip_runtime.h>

typedef __bf16 bf16_t;
typedef __attribute__((ext_vector_type(16))) __bf16 v16bf;
typedef __attribute__((ext_vector_type(8)))  float  v8f;
typedef __attribute__((ext_vector_type(4)))  unsigned int v4u;
typedef __attribute__((ext_vector_type(8)))  int v8i;
typedef __attribute__((ext_vector_type(4)))  int v4i;

#define K_CODES 1024
#define DIM     256
#define N_TOT   32768            // 32 * 32 * 32
#define ROWS_PB 128              // rows per block (divides H*W=1024 -> one image per block)
#define LDS_PITCH 264            // 256 + 8 bf16 pad: 528B row stride -> 4-bank lane stagger
#define B_PITCH_B 528            // B tile LDS row pitch (512B data + 16B TDM pad)
#define NCHUNK   (K_CODES / 16)  // 64 k-chunks of 16 codes

#if __has_builtin(__builtin_amdgcn_tensor_load_to_lds)
#define USE_TDM 1
#endif

// ---------------------------------------------------------------------------
// Kernel 0: codebook prep — emb f32 -> bf16 (WMMA B operand) + |e_k|^2 in f32
// ---------------------------------------------------------------------------
__global__ void __launch_bounds__(256)
prep_emb(const float* __restrict__ emb, bf16_t* __restrict__ emb_bf,
         float* __restrict__ bnorm) {
    const int k = blockIdx.x;
    const int t = threadIdx.x;
    const float v = emb[(size_t)k * DIM + t];
    emb_bf[(size_t)k * DIM + t] = (bf16_t)v;
    float s = v * v;
    #pragma unroll
    for (int m = 16; m >= 1; m >>= 1) s += __shfl_xor(s, m, 32);
    __shared__ float red[8];
    if ((t & 31) == 0) red[t >> 5] = s;
    __syncthreads();
    if (t == 0) {
        float tot = 0.f;
        #pragma unroll
        for (int i = 0; i < 8; ++i) tot += red[i];
        bnorm[k] = tot;
    }
}

#ifdef USE_TDM
// TDM DMA: 16 codebook rows x 256 bf16 -> LDS, 16B pad per 512B row so the
// LDS tile has a 528B pitch (conflict-free b128 fragment reads: 2 req/bank).
__device__ __forceinline__ void tdm_load_tile(unsigned lds_off, const bf16_t* gptr) {
    const unsigned long long ga = (unsigned long long)(uintptr_t)gptr;
    v4u g0;
    g0[0] = 1u;                                   // count=1, user mode, no gather
    g0[1] = lds_off;                              // lds_addr (bytes)
    g0[2] = (unsigned)(ga & 0xffffffffull);       // global_addr[31:0]
    g0[3] = (unsigned)((ga >> 32) & 0x01ffffffu)  // global_addr[56:32]
          | (2u << 30);                           // type = 2 ("image")
    v8i g1;
    g1[0] = (int)((1u << 16)                      // data_size = 1 -> 2 bytes
          |       (1u << 20)                      // pad_enable
          |       (6u << 22)                      // pad_interval: 128 dwords (=512B row)
          |       (3u << 25));                    // pad_amount: 4 dwords (=16B)
    g1[1] = (int)(256u << 16);                    // tensor_dim0 = 256 elements
    g1[2] = (int)(16u  << 16);                    // tensor_dim1 = 16 (this slice)
    g1[3] = (int)(256u << 16);                    // tile_dim0 = 256
    g1[4] = 16;                                   // tile_dim1 = 16, tile_dim2 = 0
    g1[5] = 256;                                  // tensor_dim0_stride = 256 elements
    g1[6] = 0;
    g1[7] = 0;
    const v4i gz = {0, 0, 0, 0};
#if defined(__clang_major__) && __clang_major__ >= 23
    const v8i gz8 = {0, 0, 0, 0, 0, 0, 0, 0};
    __builtin_amdgcn_tensor_load_to_lds(g0, g1, gz, gz, gz8, 0);
#else
    __builtin_amdgcn_tensor_load_to_lds(g0, g1, gz, gz, 0);
#endif
}
#else
// Fallback staging: cooperative copy with the same padded layout.
__device__ __forceinline__ void stage_chunk(bf16_t* dst, const bf16_t* src, int tid) {
    for (int q = tid; q < 16 * 32; q += 128) {    // 16B units: 32 per 512B row
        const int row = q >> 5, c = q & 31;
        *(uint4*)((char*)dst + row * B_PITCH_B + c * 16) =
            *(const uint4*)((const char*)src + row * 512 + c * 16);
    }
}
#endif

// ---------------------------------------------------------------------------
// Kernel 1: argmin_k ( |e_k|^2 - 2 * x_n . e_k ) via bf16 WMMA GEMM.
//   128 threads (4 waves); each wave owns 32 rows (two 16-row tiles): one B
//   fragment feeds 2 WMMAs (2 independent accumulator chains). B tiles are
//   TDM-DMA'd into triple-buffered LDS (prefetch distance 2) and shared.
// ---------------------------------------------------------------------------
__global__ void __launch_bounds__(128)
argmin_kernel(const float* __restrict__ x, const bf16_t* __restrict__ emb_bf,
              const float* __restrict__ bnorm, int* __restrict__ indices) {
    __shared__ __align__(16) bf16_t ldsA[ROWS_PB][LDS_PITCH];
    __shared__ __align__(16) bf16_t ldsB[3][16 * (B_PITCH_B / 2)];
    __shared__ float ldsBn[K_CODES];

    const int tid = threadIdx.x;
    const int nb  = blockIdx.x * ROWS_PB;
    const int b   = nb >> 10;                 // all 128 rows share one image
    const int hw0 = nb & 1023;
    const float* xb = x + (size_t)b * DIM * 1024 + hw0;

    const int lane  = tid & 31;
    const int wave  = tid >> 5;
    const int rbase = wave * 32;              // 32 rows per wave
    const int r     = lane & 15;
    const int hi    = lane >> 4;

#ifdef USE_TDM
    if (wave == 0) {
        tdm_load_tile((unsigned)(uintptr_t)&ldsB[0][0], emb_bf);
        tdm_load_tile((unsigned)(uintptr_t)&ldsB[1][0], emb_bf + (size_t)16 * DIM);
    }
    __builtin_amdgcn_s_wait_tensorcnt(1);         // chunk 0 resident; chunk 1 in flight
#else
    stage_chunk(&ldsB[0][0], emb_bf, tid);
#endif

    // Stage x tile: xf[n][d] = x[b][d][hw] -> LDS[n_local][d] bf16 (coalesced on hw).
    for (int e = tid; e < ROWS_PB * DIM; e += 128) {
        const int d  = e >> 7;
        const int nl = e & 127;
        ldsA[nl][d] = (bf16_t)xb[(size_t)d * 1024 + nl];
    }
    // Stage |e_k|^2 once.
    for (int i = tid; i < K_CODES; i += 128) ldsBn[i] = bnorm[i];
    __syncthreads();

    union Frag { v16bf v; uint4 q[2]; };

    // Preload A fragments for both row tiles, all of D=256 (16 x v16bf = 128 VGPRs).
    // bf16 A 16x32 layout: lanes 0-15 row M=lane, k {0..7,16..23}; lanes 16-31 k {8..15,24..31}.
    Frag A0[8], A1[8];
    #pragma unroll
    for (int dc = 0; dc < 8; ++dc) {
        const int d0 = dc * 32 + hi * 8;
        A0[dc].q[0] = *(const uint4*)&ldsA[rbase + r][d0];
        A0[dc].q[1] = *(const uint4*)&ldsA[rbase + r][d0 + 16];
        A1[dc].q[0] = *(const uint4*)&ldsA[rbase + 16 + r][d0];
        A1[dc].q[1] = *(const uint4*)&ldsA[rbase + 16 + r][d0 + 16];
    }

    float best0[8], best1[8];
    int   bidx0[8], bidx1[8];
    #pragma unroll
    for (int j = 0; j < 8; ++j) {
        best0[j] = 3.0e38f; best1[j] = 3.0e38f; bidx0[j] = 0; bidx1[j] = 0;
    }

    const size_t rowoff = (size_t)r * B_PITCH_B + hi * 16;

    // One 16-code chunk: 16 ds_load_b128 -> 16 WMMAs (2 chains) -> argmin update.
    auto do_chunk = [&](int it, int bufidx) __attribute__((always_inline)) {
        const char* Bbase = (const char*)&ldsB[bufidx][0];
        Frag Bf[2];
        Bf[0].q[0] = *(const uint4*)(Bbase + rowoff);
        Bf[0].q[1] = *(const uint4*)(Bbase + rowoff + 32);
        v8f acc0 = {}, acc1 = {};
        #pragma unroll
        for (int dc = 0; dc < 8; ++dc) {
            if (dc < 7) {                 // even/odd parity: load dc+1 ahead of dc's WMMAs
                const char* bp = Bbase + rowoff + (size_t)(dc + 1) * 64;
                Bf[(dc + 1) & 1].q[0] = *(const uint4*)bp;
                Bf[(dc + 1) & 1].q[1] = *(const uint4*)(bp + 32);
            }
            acc0 = __builtin_amdgcn_wmma_f32_16x16x32_bf16(
                       false, A0[dc].v, false, Bf[dc & 1].v, (short)0, acc0, false, false);
            acc1 = __builtin_amdgcn_wmma_f32_16x16x32_bf16(
                       false, A1[dc].v, false, Bf[dc & 1].v, (short)0, acc1, false, false);
        }
        const int   col = it * 16 + r;
        const float bn  = ldsBn[col];
        // D layout: VGPR j -> row (j + 8*hi), col (lane%16). |x|^2 dropped (row-constant).
        // col ascends within a lane, so strict '<' == first-minimum; branchless selects.
        #pragma unroll
        for (int j = 0; j < 8; ++j) {
            const float d0 = bn - 2.0f * acc0[j];
            const bool  c0 = d0 < best0[j];
            best0[j] = c0 ? d0 : best0[j];
            bidx0[j] = c0 ? col : bidx0[j];
            const float d1 = bn - 2.0f * acc1[j];
            const bool  c1 = d1 < best1[j];
            best1[j] = c1 ? d1 : best1[j];
            bidx1[j] = c1 ? col : bidx1[j];
        }
    };

#ifdef USE_TDM
    // Steady state: unconditional wait(1); last two chunks peeled (drain, then dry).
    int buf = 0, pbuf = 2;
    for (int it = 0; it < NCHUNK - 2; ++it) {
        if (wave == 0)
            tdm_load_tile((unsigned)(uintptr_t)&ldsB[pbuf][0],
                          emb_bf + (size_t)(it + 2) * 16 * DIM);
        do_chunk(it, buf);
        __builtin_amdgcn_s_wait_tensorcnt(1);     // chunk it+1 resident (waves 1-3: no-op)
        __syncthreads();
        buf  = (buf  == 2) ? 0 : buf  + 1;
        pbuf = (pbuf == 2) ? 0 : pbuf + 1;
    }
    do_chunk(NCHUNK - 2, buf);
    __builtin_amdgcn_s_wait_tensorcnt(0);         // drain: last chunk resident
    __syncthreads();
    buf = (buf == 2) ? 0 : buf + 1;
    do_chunk(NCHUNK - 1, buf);
#else
    int buf = 0, pbuf = 1;
    for (int it = 0; it < NCHUNK; ++it) {
        do_chunk(it, buf);
        __syncthreads();
        if (it + 1 < NCHUNK)
            stage_chunk(&ldsB[pbuf][0], emb_bf + (size_t)(it + 1) * 16 * DIM, tid);
        __syncthreads();
        buf  = (buf  == 2) ? 0 : buf  + 1;
        pbuf = (pbuf == 2) ? 0 : pbuf + 1;
    }
#endif

    // Reduce (min,argmin) across the 16 lanes of each half-wave; xor stays in-half.
    // Equality tie-break (lower index wins) needed only across lanes.
    #pragma unroll
    for (int t = 0; t < 2; ++t) {
        float* best = t ? best1 : best0;
        int*   bidx = t ? bidx1 : bidx0;
        #pragma unroll
        for (int j = 0; j < 8; ++j) {
            float v  = best[j];
            int   id = bidx[j];
            #pragma unroll
            for (int m = 1; m < 16; m <<= 1) {
                const float ov = __shfl_xor(v,  m, 32);
                const int   oi = __shfl_xor(id, m, 32);
                const bool  c  = (ov < v) | ((ov == v) & (oi < id));
                v  = c ? ov : v;
                id = c ? oi : id;
            }
            if (r == 0) indices[nb + rbase + t * 16 + j + hi * 8] = id;
        }
    }
}

// ---------------------------------------------------------------------------
// Kernel 2: out = emb[idx] (straight-through value == q), loss = 1.25*mean((x-q)^2)
// ---------------------------------------------------------------------------
__global__ void __launch_bounds__(256)
gather_kernel(const float* __restrict__ x, const float* __restrict__ emb,
              const int* __restrict__ indices, float* __restrict__ out,
              float* __restrict__ loss) {
    __shared__ int sidx[64];
    const int tid = threadIdx.x;
    const int nb  = blockIdx.x * 64;
    if (tid < 64) sidx[tid] = indices[nb + tid];
    __syncthreads();

    const int b   = nb >> 10;
    const int hw0 = nb & 1023;
    const size_t base = (size_t)b * DIM * 1024 + hw0;

    float partial = 0.f;
    for (int e = tid; e < 64 * DIM; e += 256) {
        const int d  = e >> 6;
        const int nl = e & 63;
        const float q  = emb[(size_t)sidx[nl] * DIM + d];   // emb hot in L2 (1 MB)
        const size_t off = base + (size_t)d * 1024 + nl;    // coalesced along hw
        const float xv = x[off];
        out[off] = q;
        const float diff = xv - q;
        partial += diff * diff;
    }

    #pragma unroll
    for (int m = 16; m >= 1; m >>= 1) partial += __shfl_xor(partial, m, 32);
    __shared__ float red[8];
    if ((tid & 31) == 0) red[tid >> 5] = partial;
    __syncthreads();
    if (tid == 0) {
        float tot = 0.f;
        #pragma unroll
        for (int i = 0; i < 8; ++i) tot += red[i];
        atomicAdd(loss, tot * (1.25f / (float)((size_t)N_TOT * DIM)));
    }
}

// ---------------------------------------------------------------------------
extern "C" void kernel_launch(void* const* d_in, const int* in_sizes, int n_in,
                              void* d_out, int out_size, void* d_ws, size_t ws_size,
                              hipStream_t stream) {
    const float* x   = (const float*)d_in[0];   // (32, 256, 32, 32) f32
    const float* emb = (const float*)d_in[1];   // (1024, 256) f32

    // d_out: out f32 [8388608], indices i32 [32768] (bit-exact), loss f32 [1]
    float* out     = (float*)d_out;
    int*   indices = (int*)((float*)d_out + (size_t)N_TOT * DIM);
    float* loss    = (float*)d_out + (size_t)N_TOT * DIM + N_TOT;

    bf16_t* emb_bf = (bf16_t*)d_ws;                                  // 512 KB
    float*  bnorm  = (float*)((char*)d_ws + (size_t)K_CODES * DIM * sizeof(bf16_t));

    prep_emb<<<K_CODES, 256, 0, stream>>>(emb, emb_bf, bnorm);
    argmin_kernel<<<N_TOT / ROWS_PB, 128, 0, stream>>>(x, emb_bf, bnorm, indices);
    hipMemsetAsync(loss, 0, sizeof(float), stream);
    gather_kernel<<<N_TOT / 64, 256, 0, stream>>>(x, emb, indices, out, loss);
}
